// BitSelfAttention_37125697307135
// MI455X (gfx1250) — 16-bit and 8-bit matrix operand order repaired and checked by reading, compiled, not run on hardware
//
#include <hip/hip_runtime.h>
#include <hip/hip_bf16.h>

typedef __attribute__((ext_vector_type(8)))  int      v8i;
typedef __attribute__((ext_vector_type(8)))  float    v8f;
typedef __attribute__((ext_vector_type(16))) _Float16 v16h;

union V16 { int i[8]; _Float16 hh[16]; v16h h; };

#define EPSQ 1e-5f
#define NEGBIG -3.0e38f

__global__ void __launch_bounds__(32)
init_zero_kernel(float* p, int n) {
    int i = threadIdx.x;
    if (i < n) p[i] = 0.f;
}

__global__ void __launch_bounds__(256)
wabs_sum_kernel(const float* __restrict__ w, float* __restrict__ sum, int n) {
    float acc = 0.f;
    for (int i = blockIdx.x * blockDim.x + threadIdx.x; i < n; i += gridDim.x * blockDim.x)
        acc += fabsf(w[i]);
#pragma unroll
    for (int off = 16; off; off >>= 1) acc += __shfl_xor(acc, off, 32);
    __shared__ float red[8];
    if ((threadIdx.x & 31) == 0) red[threadIdx.x >> 5] = acc;
    __syncthreads();
    if (threadIdx.x == 0) {
        float t = 0.f;
#pragma unroll
        for (int i = 0; i < 8; ++i) t += red[i];
        atomicAdd(sum, t);
    }
}

__global__ void __launch_bounds__(256)
wpack_kernel(const float* __restrict__ w, const float* __restrict__ sum,
             signed char* __restrict__ wq, float* __restrict__ wsc, int n) {
    float mean = fmaxf(sum[0] / (float)n, EPSQ);
    float s = 1.0f / mean;
    for (int i = blockIdx.x * blockDim.x + threadIdx.x; i < n; i += gridDim.x * blockDim.x) {
        float t = rintf(w[i] * s);
        t = fminf(fmaxf(t, -1.f), 1.f);
        wq[i] = (signed char)t;
    }
    if (blockIdx.x == 0 && threadIdx.x == 0) wsc[0] = mean;
}

__global__ void __launch_bounds__(256)
act_quant_kernel(const float* __restrict__ x, signed char* __restrict__ xq,
                 float* __restrict__ sc, int C) {
    const int row = blockIdx.x;
    const float* xr = x + (size_t)row * C;
    float amax = 0.f;
    for (int i = threadIdx.x; i < C; i += 256) amax = fmaxf(amax, fabsf(xr[i]));
#pragma unroll
    for (int off = 16; off; off >>= 1) amax = fmaxf(amax, __shfl_xor(amax, off, 32));
    __shared__ float red[8];
    if ((threadIdx.x & 31) == 0) red[threadIdx.x >> 5] = amax;
    __syncthreads();
    amax = red[0];
#pragma unroll
    for (int i = 1; i < 8; ++i) amax = fmaxf(amax, red[i]);
    amax = fmaxf(amax, EPSQ);
    float s = 127.0f / amax;
    signed char* xo = xq + (size_t)row * C;
    for (int i = threadIdx.x; i < C; i += 256) {
        float t = rintf(xr[i] * s);
        t = fminf(fmaxf(t, -128.f), 127.f);
        xo[i] = (signed char)t;
    }
    if (threadIdx.x == 0) sc[row] = amax / 127.0f;
}

__global__ void __launch_bounds__(256)
bitgemm_kernel(const signed char* __restrict__ Aq, const float* __restrict__ as,
               const signed char* __restrict__ Bq, const float* __restrict__ wsc,
               float* __restrict__ outF, _Float16* __restrict__ outH,
               const float* __restrict__ bias,
               int M, int N, int K, int mode, int H, int T, int hd) {
    const int lane = threadIdx.x & 31;
    const int wave = threadIdx.x >> 5;
    const int half = lane >> 4;
    const int l15  = lane & 15;

    const int nBlk = N >> 9;
    const int tm   = (blockIdx.x / nBlk) << 4;
    const int tn0  = (blockIdx.x % nBlk) * 512 + wave * 64;

    __shared__ signed char abuf[2][16 * 128];

    const int srow = threadIdx.x >> 4;
    const int sseg = threadIdx.x & 15;
    const signed char* agsrc = Aq + (size_t)(tm + srow) * K + sseg * 8;

    v8i acc[4] = {};

    {
        unsigned ldsoff = (unsigned)(size_t)&abuf[0][srow * 128 + sseg * 8];
        asm volatile("global_load_async_to_lds_b64 %0, %1, off"
                     :: "v"(ldsoff), "v"(agsrc) : "memory");
    }

    const int nch = K >> 7;
    for (int kc = 0; kc < nch; ++kc) {
        if (kc + 1 < nch) {
            unsigned ldsoff = (unsigned)(size_t)&abuf[(kc + 1) & 1][srow * 128 + sseg * 8];
            const signed char* g = agsrc + ((kc + 1) << 7);
            asm volatile("global_load_async_to_lds_b64 %0, %1, off"
                         :: "v"(ldsoff), "v"(g) : "memory");
            asm volatile("s_wait_asynccnt 0x1" ::: "memory");
        } else {
            asm volatile("s_wait_asynccnt 0x0" ::: "memory");
        }
        __syncthreads();

        const signed char* ab = &abuf[kc & 1][0];
#pragma unroll
        for (int ks = 0; ks < 2; ++ks) {
            v8i a;
#pragma unroll
            for (int d = 0; d < 8; ++d)
                a[d] = *(const int*)(ab + l15 * 128 + ks * 64 + half * 8 +
                                     ((d >> 1) * 16 + (d & 1) * 4));
#pragma unroll
            for (int t = 0; t < 4; ++t) {
                const signed char* bp = Bq + (size_t)(tn0 + t * 16 + l15) * K +
                                        (kc << 7) + ks * 64 + half * 8;
                v8i b;
#pragma unroll
                for (int d = 0; d < 8; ++d)
                    b[d] = *(const int*)(bp + ((d >> 1) * 16 + (d & 1) * 4));
                acc[t] = __builtin_amdgcn_wmma_i32_16x16x64_iu8(true, a, true, b,
                                                                acc[t], false, false);
            }
        }
        __syncthreads();
    }

    const float ws = wsc[0];
#pragma unroll
    for (int t = 0; t < 4; ++t) {
#pragma unroll
        for (int r = 0; r < 8; ++r) {
            const int row = tm + r + half * 8;
            const int col = tn0 + t * 16 + l15;
            float val = (float)acc[t][r] * as[row] * ws;
            if (mode == 0) {
                outF[(size_t)row * N + col] = val + bias[col];
            } else {
                const int b_ = row / T, t_ = row % T;
                const int h_ = col / hd, d_ = col % hd;
                outH[(((size_t)b_ * H + h_) * T + t_) * hd + d_] = (_Float16)val;
            }
        }
    }
}

__global__ void __launch_bounds__(32)
flash_kernel(const _Float16* __restrict__ qm, const _Float16* __restrict__ km,
             const _Float16* __restrict__ vm, float* __restrict__ o,
             int H, int T, int hd, float scale) {
    const int lane = threadIdx.x & 31;
    const int half = lane >> 4;
    const int l15  = lane & 15;
    const int QT = T >> 4;
    const int qt = blockIdx.x % QT;
    const int bh = blockIdx.x / QT;
    const int b_ = bh / H, h_ = bh % H;
    const int C = H * hd;

    const _Float16* qb = qm + ((size_t)bh * T + qt * 16) * hd;
    const _Float16* kb = km + (size_t)bh * T * hd;
    const _Float16* vb = vm + (size_t)bh * T * hd;

    V16 qa[2];
#pragma unroll
    for (int kk = 0; kk < 2; ++kk) {
        const _Float16* qp = qb + (size_t)l15 * hd + kk * 32 + half * 8;
#pragma unroll
        for (int dd = 0; dd < 8; ++dd)
            qa[kk].i[dd] = *(const int*)(qp + (dd >> 2) * 16 + (dd & 3) * 2);
    }

    float mrow[8], lrow[8];
    v8f oacc[4] = {};
#pragma unroll
    for (int r = 0; r < 8; ++r) { mrow[r] = NEGBIG; lrow[r] = 0.f; }

    __shared__ _Float16 pbuf[16 * 32];

    const int ktmax = (qt * 16 + 15) >> 5;
    for (int kt = 0; kt <= ktmax; ++kt) {
        v8f s[2];
#pragma unroll
        for (int nc = 0; nc < 2; ++nc) {
            v8f sacc = {};
#pragma unroll
            for (int kk = 0; kk < 2; ++kk) {
                V16 ub;
                const _Float16* kp = kb + (size_t)(kt * 32 + nc * 16 + l15) * hd + kk * 32 + half * 8;
#pragma unroll
                for (int dd = 0; dd < 8; ++dd) ub.i[dd] = *(const int*)(kp + (dd >> 2) * 16 + (dd & 3) * 2);
                sacc = __builtin_amdgcn_wmma_f32_16x16x32_f16(false, qa[kk].h, false, ub.h,
                                                              (short)0, sacc, false, false);
            }
            s[nc] = sacc;
        }

#pragma unroll
        for (int r = 0; r < 8; ++r) {
            const int qi = qt * 16 + r + half * 8;
            float mr = NEGBIG;
#pragma unroll
            for (int nc = 0; nc < 2; ++nc) {
                const int kj = kt * 32 + nc * 16 + l15;
                float val = s[nc][r] * scale;
                val = (kj <= qi) ? val : NEGBIG;
                s[nc][r] = val;
                mr = fmaxf(mr, val);
            }
#pragma unroll
            for (int off = 1; off < 16; off <<= 1) mr = fmaxf(mr, __shfl_xor(mr, off, 32));
            const float mnew = fmaxf(mrow[r], mr);
            const float corr = __expf(mrow[r] - mnew);
            float pl = 0.f;
#pragma unroll
            for (int nc = 0; nc < 2; ++nc) {
                float p = __expf(s[nc][r] - mnew);
                s[nc][r] = p;
                pl += p;
            }
#pragma unroll
            for (int off = 1; off < 16; off <<= 1) pl += __shfl_xor(pl, off, 32);
            lrow[r] = lrow[r] * corr + pl;
            mrow[r] = mnew;
#pragma unroll
            for (int c = 0; c < 4; ++c) oacc[c][r] *= corr;
        }

#pragma unroll
        for (int r = 0; r < 8; ++r)
#pragma unroll
            for (int nc = 0; nc < 2; ++nc)
                pbuf[(r + half * 8) * 32 + nc * 16 + l15] = (_Float16)s[nc][r];
        __syncthreads();
        V16 pa;
#pragma unroll
        for (int dd = 0; dd < 8; ++dd)
            pa.i[dd] = *(const int*)&pbuf[(size_t)l15 * 32 + (dd >> 2) * 16 + (dd & 3) * 2 + half * 8];
        __syncthreads();

#pragma unroll
        for (int dc = 0; dc < 4; ++dc) {
            V16 ub;
            const _Float16* vp = vb + (size_t)(kt * 32 + half * 8) * hd + dc * 16 + l15;
#pragma unroll
            for (int j = 0; j < 16; ++j) ub.hh[j] = vp[(size_t)((j >> 3) * 16 + (j & 7)) * hd];
            oacc[dc] = __builtin_amdgcn_wmma_f32_16x16x32_f16(false, pa.h, false, ub.h,
                                                              (short)0, oacc[dc], false, false);
        }
    }

#pragma unroll
    for (int r = 0; r < 8; ++r) {
        const int t_ = qt * 16 + r + half * 8;
        const float inv = 1.0f / lrow[r];
#pragma unroll
        for (int dc = 0; dc < 4; ++dc)
            o[((size_t)b_ * T + t_) * C + h_ * hd + dc * 16 + l15] = oacc[dc][r] * inv;
    }
}

extern "C" void kernel_launch(void* const* d_in, const int* in_sizes, int n_in,
                              void* d_out, int out_size, void* d_ws, size_t ws_size,
                              hipStream_t stream) {
    (void)in_sizes; (void)n_in; (void)out_size; (void)ws_size;
    const float* x  = (const float*)d_in[0];
    const float* Wlist[4] = { (const float*)d_in[1], (const float*)d_in[2],
                              (const float*)d_in[3], (const float*)d_in[4] };
    const float* bp = (const float*)d_in[5];

    const int Bn = 64, T = 256, C = 1024, H = 16, hd = 64;
    const int M = Bn * T;
    float* out = (float*)d_out;

    char* ws = (char*)d_ws;
    size_t off = 0;
    auto carve = [&](size_t bytes) -> char* {
        char* p = ws + off;
        off = (off + bytes + 255) & ~(size_t)255;
        return p;
    };
    signed char* xq = (signed char*)carve((size_t)M * C);
    signed char* yq = (signed char*)carve((size_t)M * C);
    signed char* wq8[4];
    for (int i = 0; i < 4; ++i) wq8[i] = (signed char*)carve((size_t)C * C);
    float* ascale = (float*)carve((size_t)M * 4);
    float* yscale = (float*)carve((size_t)M * 4);
    float* wsum   = (float*)carve(16);
    float* wsc    = (float*)carve(16);
    _Float16* qh  = (_Float16*)carve((size_t)M * C * 2);
    _Float16* kh  = (_Float16*)carve((size_t)M * C * 2);
    _Float16* vh  = (_Float16*)carve((size_t)M * C * 2);
    float* obuf   = (float*)carve((size_t)M * C * 4);

    init_zero_kernel<<<1, 32, 0, stream>>>(wsum, 4);
    for (int i = 0; i < 4; ++i)
        wabs_sum_kernel<<<512, 256, 0, stream>>>(Wlist[i], wsum + i, C * C);
    for (int i = 0; i < 4; ++i)
        wpack_kernel<<<512, 256, 0, stream>>>(Wlist[i], wsum + i, wq8[i], wsc + i, C * C);

    act_quant_kernel<<<M, 256, 0, stream>>>(x, xq, ascale, C);

    const int gemmBlocks = (M / 16) * (C / 512);
    bitgemm_kernel<<<gemmBlocks, 256, 0, stream>>>(xq, ascale, wq8[0], wsc + 0,
                                                   nullptr, qh, nullptr, M, C, C, 1, H, T, hd);
    bitgemm_kernel<<<gemmBlocks, 256, 0, stream>>>(xq, ascale, wq8[1], wsc + 1,
                                                   nullptr, kh, nullptr, M, C, C, 1, H, T, hd);
    bitgemm_kernel<<<gemmBlocks, 256, 0, stream>>>(xq, ascale, wq8[2], wsc + 2,
                                                   nullptr, vh, nullptr, M, C, C, 1, H, T, hd);

    flash_kernel<<<Bn * H * (T / 16), 32, 0, stream>>>(qh, kh, vh, obuf, H, T, hd, 0.125f);

    act_quant_kernel<<<M, 256, 0, stream>>>(obuf, yq, yscale, C);
    bitgemm_kernel<<<gemmBlocks, 256, 0, stream>>>(yq, yscale, wq8[3], wsc + 3,
                                                   out, nullptr, bp, M, C, C, 0, H, T, hd);
}
